// CTC_75368086110225
// MI455X (gfx1250) — matrix operand order repaired and checked by reading, not compiled, not run on hardware
//
#include <hip/hip_runtime.h>
#include <hip/hip_bf16.h>
#include <math.h>

#define B_ 16
#define T_ 512
#define E_ 768
#define V_ 10000
#define L_ 100
#define S_ 201
#define NEGV (-1e30f)

typedef __attribute__((ext_vector_type(16))) __bf16 bf16x16;
typedef __attribute__((ext_vector_type(8)))  float  f32x8;
typedef __attribute__((ext_vector_type(4)))  unsigned int uint4v;

union FragBF { uint4v u[2]; bf16x16 v; };

__device__ __forceinline__ unsigned short f2bf(float f) {
  unsigned int u = __float_as_uint(f);
  u += 0x7FFFu + ((u >> 16) & 1u);          // round-to-nearest-even
  return (unsigned short)(u >> 16);
}
__device__ __forceinline__ float bf2f(unsigned short h) {
  return __uint_as_float(((unsigned int)h) << 16);
}

// ---------------- 1. extended labels + skip flags ----------------
__global__ void k_prep_ext(const int* __restrict__ ys,
                           int* __restrict__ ext, int* __restrict__ skip) {
  int idx = blockIdx.x * blockDim.x + threadIdx.x;
  if (idx >= B_ * S_) return;
  int b = idx / S_, s = idx % S_;
  int e = 0;
  if (s & 1) { int lab = ys[b * L_ + ((s - 1) >> 1)]; e = lab < 0 ? 0 : lab; }
  int em2 = 0;
  if (s >= 2 && ((s - 2) & 1)) { int lab = ys[b * L_ + ((s - 3) >> 1)]; em2 = lab < 0 ? 0 : lab; }
  ext[idx]  = e;
  skip[idx] = (s >= 2 && e != 0 && e != em2) ? 1 : 0;
}

// ---------------- 2. fp32 -> bf16 hi/lo split (row-major copy) ----------------
__global__ void k_cvt_split(const float* __restrict__ x,
                            unsigned short* __restrict__ hi,
                            unsigned short* __restrict__ lo, int n) {
  int i = blockIdx.x * blockDim.x + threadIdx.x;
  if (i >= n) return;
  float v = x[i];
  unsigned short h = f2bf(v);
  hi[i] = h;
  lo[i] = f2bf(v - bf2f(h));
}

// ---------------- 3. W [E,V] fp32 -> Wt [V,E] bf16 hi/lo (LDS transpose) ----------------
__global__ void k_cvt_w(const float* __restrict__ W,
                        unsigned short* __restrict__ hi,
                        unsigned short* __restrict__ lo) {
  __shared__ float tile[32][33];
  int tx = threadIdx.x, ty = threadIdx.y;
  int v0 = blockIdx.x * 32, e0 = blockIdx.y * 32;
  for (int r = ty; r < 32; r += 8) {
    int e = e0 + r, v = v0 + tx;
    tile[r][tx] = (v < V_) ? W[(size_t)e * V_ + v] : 0.f;
  }
  __syncthreads();
  for (int r = ty; r < 32; r += 8) {
    int v = v0 + r, e = e0 + tx;
    if (v < V_) {
      float x = tile[tx][r];
      unsigned short h = f2bf(x);
      hi[(size_t)v * E_ + e] = h;
      lo[(size_t)v * E_ + e] = f2bf(x - bf2f(h));
    }
  }
}

// ---------------- 4. fused GEMM (bf16x3 WMMA) + online row logsumexp ----------------
// One WG = 16 rows, loops over all 625 N-tiles (8 waves strided).
__global__ void __launch_bounds__(256) k_gemm_lse(
    const unsigned short* __restrict__ Ahi, const unsigned short* __restrict__ Alo,
    const unsigned short* __restrict__ Bthi, const unsigned short* __restrict__ Btlo,
    const float* __restrict__ bias, float* __restrict__ denom) {
  __shared__ __align__(16) unsigned short AsH[16 * E_];
  __shared__ __align__(16) unsigned short AsL[16 * E_];
  __shared__ float Wm[8][16];
  __shared__ float Wsum[8][16];

  const int tid  = threadIdx.x;
  const int row0 = blockIdx.x * 16;

  { // A tile -> LDS (contiguous 16 rows, vectorized)
    const uint4v* gh = (const uint4v*)(Ahi + (size_t)row0 * E_);
    const uint4v* gl = (const uint4v*)(Alo + (size_t)row0 * E_);
    uint4v* sh = (uint4v*)AsH;
    uint4v* sl = (uint4v*)AsL;
    const int nvec = 16 * E_ / 8;
    for (int i = tid; i < nvec; i += 256) { sh[i] = gh[i]; sl[i] = gl[i]; }
  }
  __syncthreads();

  const int wave = tid >> 5, lane = tid & 31;
  const int half = lane >> 4, col = lane & 15;
  const int m = lane & 15;
  const unsigned short* arow_h = AsH + m * E_;
  const unsigned short* arow_l = AsL + m * E_;

  float mrun[8], srun[8];
#pragma unroll
  for (int i = 0; i < 8; i++) { mrun[i] = -3.0e38f; srun[i] = 0.f; }

  for (int j = wave; j < V_ / 16; j += 8) {
    const int n = j * 16 + col;
    const unsigned short* bcol_h = Bthi + (size_t)n * E_;
    const unsigned short* bcol_l = Btlo + (size_t)n * E_;
    f32x8 c = {};
#pragma unroll 4
    for (int kt = 0; kt < E_ / 32; kt++) {
      const int ka = kt * 32 + half * 8;    // A: lanes<16 carry K {0..7,16..23}
      const int kb = kt * 32 + half * 8;    // B: lanes<16 carry K {0..7,16..23}
      FragBF aH, aL, bH, bL;
      aH.u[0] = *(const uint4v*)(arow_h + ka);
      aH.u[1] = *(const uint4v*)(arow_h + ka + 16);
      aL.u[0] = *(const uint4v*)(arow_l + ka);
      aL.u[1] = *(const uint4v*)(arow_l + ka + 16);
      bH.u[0] = *(const uint4v*)(bcol_h + kb);
      bH.u[1] = *(const uint4v*)(bcol_h + kb + 16);
      bL.u[0] = *(const uint4v*)(bcol_l + kb);
      bL.u[1] = *(const uint4v*)(bcol_l + kb + 16);
      // fp32-accurate product via bf16 split: hi*hi + hi*lo + lo*hi
      c = __builtin_amdgcn_wmma_f32_16x16x32_bf16(false, aH.v, false, bH.v, (short)0, c, false, false);
      c = __builtin_amdgcn_wmma_f32_16x16x32_bf16(false, aH.v, false, bL.v, (short)0, c, false, false);
      c = __builtin_amdgcn_wmma_f32_16x16x32_bf16(false, aL.v, false, bH.v, (short)0, c, false, false);
    }
    const float bn = bias[n];
#pragma unroll
    for (int i = 0; i < 8; i++) {
      float x = c[i] + bn;
      float tmax = x;
      tmax = fmaxf(tmax, __shfl_xor(tmax, 1, 16));
      tmax = fmaxf(tmax, __shfl_xor(tmax, 2, 16));
      tmax = fmaxf(tmax, __shfl_xor(tmax, 4, 16));
      tmax = fmaxf(tmax, __shfl_xor(tmax, 8, 16));
      float mnew = fmaxf(mrun[i], tmax);
      float e = expf(x - mnew);
      e += __shfl_xor(e, 1, 16);
      e += __shfl_xor(e, 2, 16);
      e += __shfl_xor(e, 4, 16);
      e += __shfl_xor(e, 8, 16);
      srun[i] = srun[i] * expf(mrun[i] - mnew) + e;
      mrun[i] = mnew;
    }
  }
  if (col == 0) {
#pragma unroll
    for (int i = 0; i < 8; i++) {       // C layout: VGPR i = row i (+8 for upper half)
      Wm[wave][half * 8 + i]   = mrun[i];
      Wsum[wave][half * 8 + i] = srun[i];
    }
  }
  __syncthreads();
  if (tid < 16) {
    float M = -3.0e38f;
    for (int w = 0; w < 8; w++) M = fmaxf(M, Wm[w][tid]);
    float S = 0.f;
    for (int w = 0; w < 8; w++) S += Wsum[w][tid] * expf(Wm[w][tid] - M);
    denom[row0 + tid] = M + logf(S);
  }
}

// ---------------- 5. gather numerator logits, subtract denom -> lp[b][t][s] ----------------
__global__ void __launch_bounds__(256) k_gather_lp(
    const float* __restrict__ hs, const float* __restrict__ W,
    const float* __restrict__ bias, const int* __restrict__ ext,
    const float* __restrict__ denom, float* __restrict__ lp) {
  const int bs = blockIdx.x;
  const int b = bs / S_, s = bs % S_;
  const int cls = ext[bs];
  __shared__ float wcol[E_];
  const int tid = threadIdx.x;
  for (int i = tid; i < E_; i += 256) wcol[i] = W[(size_t)i * V_ + cls];
  __syncthreads();
  const float bc = bias[cls];
  const int wave = tid >> 5, lane = tid & 31;
  for (int t = wave; t < T_; t += 8) {
    const float* hrow = hs + (size_t)(b * T_ + t) * E_;
    float acc = 0.f;
#pragma unroll
    for (int i = 0; i < E_ / 32; i++) {
      int e = lane + i * 32;
      acc += hrow[e] * wcol[e];
    }
#pragma unroll
    for (int off = 16; off >= 1; off >>= 1) acc += __shfl_xor(acc, off, 32);
    if (lane == 0)
      lp[(size_t)(b * T_ + t) * S_ + s] = acc + bc - denom[b * T_ + t];
  }
}

// ---------------- 6. CTC forward recursion (one WG per batch) ----------------
__global__ void __launch_bounds__(256) k_ctc(
    const float* __restrict__ lp, const int* __restrict__ skip,
    const int* __restrict__ hlens, const int* __restrict__ olens,
    float* __restrict__ lossb) {
  const int b = blockIdx.x, tid = threadIdx.x;
  __shared__ float al[S_];
  __shared__ int sk[S_];
  for (int i = tid; i < S_; i += 256) { al[i] = NEGV; sk[i] = skip[b * S_ + i]; }
  __syncthreads();
  if (tid == 0) al[0] = lp[(size_t)(b * T_) * S_ + 0];
  if (tid == 1) al[1] = lp[(size_t)(b * T_) * S_ + 1];
  __syncthreads();
  const int ilen = hlens[b];
  for (int t = 1; t < T_; t++) {
    float nv = 0.f;
    const bool act = (t < ilen);
    if (tid < S_) {
      float a  = al[tid];
      float a1 = (tid >= 1) ? al[tid - 1] : NEGV;
      float a2 = sk[tid] ? al[tid - 2] : NEGV;
      float mm = fmaxf(a, fmaxf(a1, a2));
      float cand = mm + logf(expf(a - mm) + expf(a1 - mm) + expf(a2 - mm));
      nv = cand + lp[(size_t)(b * T_ + t) * S_ + tid];
      if (!act) nv = a;
    }
    __syncthreads();
    if (tid < S_) al[tid] = nv;
    __syncthreads();
  }
  if (tid == 0) {
    int sl = 2 * olens[b];
    float lb = al[sl], ll = al[sl - 1];
    float mm = fmaxf(lb, ll);
    lossb[b] = -(mm + logf(expf(lb - mm) + expf(ll - mm)));
  }
}

// ---------------- 7. final mean ----------------
__global__ void k_final(const float* __restrict__ lossb, float* __restrict__ out) {
  if (threadIdx.x == 0) {
    float s = 0.f;
    for (int b = 0; b < B_; b++) s += lossb[b];
    out[0] = s / (float)B_;
  }
}

extern "C" void kernel_launch(void* const* d_in, const int* in_sizes, int n_in,
                              void* d_out, int out_size, void* d_ws, size_t ws_size,
                              hipStream_t stream) {
  (void)in_sizes; (void)n_in; (void)out_size; (void)ws_size;
  const float* hs    = (const float*)d_in[0];
  const int*   hlens = (const int*)d_in[1];
  const int*   ys    = (const int*)d_in[2];
  const int*   olens = (const int*)d_in[3];
  const float* W     = (const float*)d_in[4];
  const float* bias  = (const float*)d_in[5];
  float* out = (float*)d_out;

  char* ws = (char*)d_ws;
  size_t off = 0;
  auto alloc = [&](size_t bytes) -> void* {
    void* p = ws + off;
    off = (off + bytes + 255) & ~(size_t)255;
    return p;
  };
  unsigned short* Ahi  = (unsigned short*)alloc((size_t)B_ * T_ * E_ * 2);
  unsigned short* Alo  = (unsigned short*)alloc((size_t)B_ * T_ * E_ * 2);
  unsigned short* Wthi = (unsigned short*)alloc((size_t)V_ * E_ * 2);
  unsigned short* Wtlo = (unsigned short*)alloc((size_t)V_ * E_ * 2);
  float* denom = (float*)alloc((size_t)B_ * T_ * 4);
  float* lp    = (float*)alloc((size_t)B_ * T_ * S_ * 4);
  int*   ext   = (int*)alloc((size_t)B_ * S_ * 4);
  int*   skip  = (int*)alloc((size_t)B_ * S_ * 4);
  float* lossb = (float*)alloc((size_t)B_ * 4);

  k_prep_ext<<<(B_ * S_ + 255) / 256, 256, 0, stream>>>(ys, ext, skip);

  const int nA = B_ * T_ * E_;
  k_cvt_split<<<(nA + 255) / 256, 256, 0, stream>>>(hs, Ahi, Alo, nA);

  k_cvt_w<<<dim3((V_ + 31) / 32, E_ / 32), dim3(32, 8), 0, stream>>>(W, Wthi, Wtlo);

  k_gemm_lse<<<(B_ * T_) / 16, 256, 0, stream>>>(Ahi, Alo, Wthi, Wtlo, bias, denom);

  k_gather_lp<<<B_ * S_, 256, 0, stream>>>(hs, W, bias, ext, denom, lp);

  k_ctc<<<B_, 256, 0, stream>>>(lp, skip, hlens, olens, lossb);

  k_final<<<1, 32, 0, stream>>>(lossb, out);
}
